// LSTM_51651276702190
// MI455X (gfx1250) — matrix operand order repaired and checked by reading, not compiled, not run on hardware
//
#include <hip/hip_runtime.h>

typedef __attribute__((ext_vector_type(16))) _Float16 v16h;
typedef __attribute__((ext_vector_type(8)))  float    v8f;

#define SLEN 512
#define BSZ  32
#define ROWS (SLEN*BSZ)      // 16384
#define WDIM 300
#define DPAD 320             // 311 padded to 10 chunks of 32
#define HDIM 64
#define G4H  256
#define NCLS 13

// ---------------- fragment loaders ----------------
// A fragment: 16x32 (MxK) f16 from row-major matrix, lane L holds row L%16,
// elems 0..7 = k[kb..kb+7], elems 8..15 = k[kb+16..kb+23], kb = (L>=16)*8
__device__ __forceinline__ v16h load_a16x32(const _Float16* base, int ld) {
  int lane = threadIdx.x & 31;
  const _Float16* p = base + (lane & 15) * ld + ((lane >> 4) << 3);
  v16h r;
  *(float4*)(&r)       = *(const float4*)(p);
  *((float4*)(&r) + 1) = *(const float4*)(p + 16);
  return r;
}
// Same A fragment but converting from f32 source
__device__ __forceinline__ v16h load_a16x32_f32(const float* base, int ld) {
  int lane = threadIdx.x & 31;
  const float* p = base + (lane & 15) * ld + ((lane >> 4) << 3);
  v16h r;
#pragma unroll
  for (int i = 0; i < 8; ++i) r[i] = (_Float16)p[i];
#pragma unroll
  for (int i = 0; i < 8; ++i) r[8 + i] = (_Float16)p[16 + i];
  return r;
}
// B fragment for computing X @ W^T: W row-major [N][K]; lane L holds
// column n = L%16 of B (= row n of W), elems 0..7 = k[kb..kb+7], 8..15 = k[kb+16..kb+23], kb = (L>=16)*8
__device__ __forceinline__ v16h load_b32x16(const _Float16* wbase, int ld) {
  int lane = threadIdx.x & 31;
  const _Float16* p = wbase + (lane & 15) * ld + ((lane >> 4) << 3);
  v16h r;
  *(float4*)(&r)       = *(const float4*)(p);
  *((float4*)(&r) + 1) = *(const float4*)(p + 16);
  return r;
}
#define WMMA(a, b, c) __builtin_amdgcn_wmma_f32_16x16x32_f16(false, (a), false, (b), (short)0, (c), false, false)

// Fast activations on v_exp_f32 / v_rcp_f32 (exact +-1 / 0,1 saturation at inf)
__device__ __forceinline__ float sigmoidf_(float x) {
  return __builtin_amdgcn_rcpf(1.0f + __expf(-x));
}
__device__ __forceinline__ float tanhf_(float x) {
  return 1.0f - 2.0f * __builtin_amdgcn_rcpf(__expf(2.0f * x) + 1.0f);
}

// ---------------- phase 0: zero bag accumulator ----------------
__global__ void zero_f(float* __restrict__ p, int n) {
  int i = blockIdx.x * blockDim.x + threadIdx.x;
  int stride = gridDim.x * blockDim.x;
  for (; i < n; i += stride) p[i] = 0.0f;
}

// ---------------- phase 1: bag-of-words segment sum ----------------
__global__ void scatter_bow(const int* __restrict__ wids, const int* __restrict__ wseg,
                            const float* __restrict__ wtab, float* __restrict__ bags) {
  int t = blockIdx.x;                    // one word per block
  size_t wid = (size_t)wids[t];
  size_t seg = (size_t)wseg[t];
  const float* src = wtab + wid * WDIM;
  float* dst = bags + seg * WDIM;
  for (int d = threadIdx.x; d < WDIM; d += 128)
    atomicAdd(&dst[d], src[d]);
}

// ---------------- phase 2: assemble x (f16, padded to 320) ----------------
__global__ void pack_x(const int* __restrict__ spk, const int* __restrict__ da,
                       const int* __restrict__ lvl, const float* __restrict__ ulen,
                       const float* __restrict__ e_spk, const float* __restrict__ e_da,
                       const float* __restrict__ e_lvl, const float* __restrict__ bags,
                       _Float16* __restrict__ xh) {
  int row = blockIdx.x;                  // 0..16383 == s*B + b
  int t = threadIdx.x;                   // 0..63
  int sp = spk[row], dd = da[row], lv = lvl[row];
#pragma unroll
  for (int q = 0; q < 5; ++q) {
    int c = t + 64 * q;
    float v;
    if (c == 0)        v = e_spk[sp];
    else if (c < 8)    v = e_da[dd * 7 + (c - 1)];
    else if (c < 10)   v = e_lvl[lv * 2 + (c - 8)];
    else if (c == 10)  v = ulen[row];
    else if (c < 311)  v = bags[(size_t)row * WDIM + (c - 11)];
    else               v = 0.0f;
    xh[(size_t)row * DPAD + c] = (_Float16)v;
  }
}

// ---------------- phase 3: convert weights to f16 (padded) ----------------
__global__ void conv_w(const float* __restrict__ W_ih, const float* __restrict__ W_hh,
                       const float* __restrict__ W_dec,
                       _Float16* __restrict__ wih, _Float16* __restrict__ whh,
                       _Float16* __restrict__ wdec) {
  int tid = blockIdx.x * blockDim.x + threadIdx.x;
  if (tid < 256 * DPAD) {
    int n = tid / DPAD, k = tid % DPAD;
    wih[tid] = (k < 311) ? (_Float16)W_ih[n * 311 + k] : (_Float16)0.0f;
  }
  int t2 = tid - 256 * DPAD;
  if (t2 >= 0 && t2 < 256 * HDIM) whh[t2] = (_Float16)W_hh[t2];
  int t3 = t2 - 256 * HDIM;
  if (t3 >= 0 && t3 < 16 * HDIM) {
    int n = t3 / HDIM;
    wdec[t3] = (n < NCLS) ? (_Float16)W_dec[t3] : (_Float16)0.0f;
  }
}

// ---------------- phase 4: big GEMM gx = x @ W_ih^T + b_ih + b_hh ----------------
// 2048 waves; wave w: m-tile = w>>1 (16 rows), n-half = w&1 (8 n-tiles = 128 cols).
__global__ void __launch_bounds__(256, 1) gemm_gates(
    const _Float16* __restrict__ xh, const _Float16* __restrict__ wih,
    const float* __restrict__ b_ih, const float* __restrict__ b_hh,
    float* __restrict__ gx) {
  int wave = (blockIdx.x * blockDim.x + threadIdx.x) >> 5;  // 0..2047
  int lane = threadIdx.x & 31;
  int m0 = (wave >> 1) << 4;
  int nb = (wave & 1) << 3;               // first n-tile of this wave's half
  v8f acc[8];
#pragma unroll
  for (int i = 0; i < 8; ++i) acc[i] = {};
  for (int kc = 0; kc < 10; ++kc) {
    v16h a = load_a16x32(xh + (size_t)m0 * DPAD + kc * 32, DPAD);
#pragma unroll
    for (int nt = 0; nt < 8; ++nt) {
      v16h b = load_b32x16(wih + (size_t)((nb + nt) * 16) * DPAD + kc * 32, DPAD);
      acc[nt] = WMMA(a, b, acc[nt]);
    }
  }
  int r_off = (lane >> 4) << 3;
  int cs = lane & 15;
#pragma unroll
  for (int nt = 0; nt < 8; ++nt) {
    int col = (nb + nt) * 16 + cs;
    float bias = b_ih[col] + b_hh[col];
#pragma unroll
    for (int v = 0; v < 8; ++v)
      gx[(size_t)(m0 + r_off + v) * G4H + col] = acc[nt][v] + bias;
  }
}

// ---------------- phase 5: sequential LSTM, one workgroup ----------------
// Wave w: mt = w>>2 (rows mt*16..+15), jg = w&3 (j in [jg*16, jg*16+16)).
// Owns n-tiles {jg, jg+4, jg+8, jg+12} = i/f/g/o gate columns for its own
// (row, j) block -> in-register LSTM update, c-state register-resident.
// h double-buffered in LDS -> ONE barrier per step.
__global__ void __launch_bounds__(256, 1) lstm_seq(
    const float* __restrict__ gx, const _Float16* __restrict__ whh,
    const float* __restrict__ h0, const float* __restrict__ c0,
    float* __restrict__ hs, float* __restrict__ hT, float* __restrict__ cT) {
  __shared__ __align__(16) _Float16 h_buf[2][BSZ * HDIM];  // 2 x 4 KB
  int tid = threadIdx.x;
  int lane = tid & 31;
  int wave = tid >> 5;                    // 0..7
  int mt = wave >> 2;                     // 0..1 (m-tile)
  int jg = wave & 3;                      // 0..3 (j-group of 16 columns)
  int r_off = (lane >> 4) << 3;
  int cs = lane & 15;
  // init h buffer 0 from h0
  for (int i = tid; i < BSZ * HDIM; i += 256)
    h_buf[0][i] = (_Float16)h0[i];
  // register-resident cell state: lane element v <-> (row, j)
  int row0 = mt * 16 + r_off;             // + v
  int jcol = jg * 16 + cs;
  v8f c_reg;
#pragma unroll
  for (int v = 0; v < 8; ++v)
    c_reg[v] = c0[(row0 + v) * HDIM + jcol];
  // W_hh fragments for the 4 owned n-tiles x 2 k-chunks (constant over steps)
  v16h Bf[4][2];
#pragma unroll
  for (int g = 0; g < 4; ++g)
#pragma unroll
    for (int kc = 0; kc < 2; ++kc)
      Bf[g][kc] = load_b32x16(whh + (size_t)((jg + 4 * g) * 16) * HDIM + kc * 32, HDIM);
  __syncthreads();
  for (int s = 0; s < SLEN; ++s) {
    const float* gxs = gx + (size_t)s * BSZ * G4H;
    const _Float16* hrd = h_buf[s & 1];
    _Float16* hwr = h_buf[(s + 1) & 1];
    // accumulators = precomputed input gates (x@W_ih^T + biases)
    v8f acc[4];
#pragma unroll
    for (int g = 0; g < 4; ++g) {
      int col = (jg + 4 * g) * 16 + cs;
#pragma unroll
      for (int v = 0; v < 8; ++v)
        acc[g][v] = gxs[(mt * 16 + r_off + v) * G4H + col];
    }
    if (s + 1 < SLEN)   // pull next step's precomputed gates toward L2/L0
      __builtin_prefetch(gx + (size_t)(s + 1) * BSZ * G4H + tid * 32, 0, 0);
    // += h @ W_hh^T for this wave's m-tile and 4 gate column-tiles
#pragma unroll
    for (int kc = 0; kc < 2; ++kc) {
      v16h a = load_a16x32(hrd + mt * 16 * HDIM + kc * 32, HDIM);
#pragma unroll
      for (int g = 0; g < 4; ++g)
        acc[g] = WMMA(a, Bf[g][kc], acc[g]);
    }
    // in-register elementwise LSTM update (acc[0]=i, [1]=f, [2]=g, [3]=o)
#pragma unroll
    for (int v = 0; v < 8; ++v) {
      float cn = sigmoidf_(acc[1][v]) * c_reg[v] +
                 sigmoidf_(acc[0][v]) * tanhf_(acc[2][v]);
      float hn = sigmoidf_(acc[3][v]) * tanhf_(cn);
      c_reg[v] = cn;
      int idx = (row0 + v) * HDIM + jcol;
      hwr[idx] = (_Float16)hn;
      hs[(size_t)s * (BSZ * HDIM) + idx] = hn;
      if (s == SLEN - 1) { hT[idx] = hn; cT[idx] = cn; }
    }
    __syncthreads();
  }
}

// ---------------- phase 6: decode = hs @ W_dec^T + b_dec ----------------
__global__ void __launch_bounds__(256, 1) decode_k(
    const float* __restrict__ hs, const _Float16* __restrict__ wdec,
    const float* __restrict__ b_dec, float* __restrict__ out) {
  int wave = (blockIdx.x * blockDim.x + threadIdx.x) >> 5;  // 0..1023 -> m-tile
  int lane = threadIdx.x & 31;
  int m0 = wave << 4;
  v8f acc = {};
#pragma unroll
  for (int kc = 0; kc < 2; ++kc) {
    v16h a = load_a16x32_f32(hs + (size_t)m0 * HDIM + kc * 32, HDIM);
    v16h b = load_b32x16(wdec + kc * 32, HDIM);
    acc = WMMA(a, b, acc);
  }
  int n = lane & 15;
  int r_off = (lane >> 4) << 3;
  if (n < NCLS) {
    float bias = b_dec[n];
#pragma unroll
    for (int v = 0; v < 8; ++v)
      out[(size_t)(m0 + r_off + v) * NCLS + n] = acc[v] + bias;
  }
}

extern "C" void kernel_launch(void* const* d_in, const int* in_sizes, int n_in,
                              void* d_out, int out_size, void* d_ws, size_t ws_size,
                              hipStream_t stream) {
  const int*   spk   = (const int*)d_in[0];
  const int*   da    = (const int*)d_in[1];
  const int*   lvl   = (const int*)d_in[2];
  const float* ulen  = (const float*)d_in[3];
  const int*   wids  = (const int*)d_in[4];
  const int*   wseg  = (const int*)d_in[5];
  const float* e_spk = (const float*)d_in[6];
  const float* e_da  = (const float*)d_in[7];
  const float* e_lvl = (const float*)d_in[8];
  const float* wtab  = (const float*)d_in[9];
  const float* W_ih  = (const float*)d_in[10];
  const float* W_hh  = (const float*)d_in[11];
  const float* b_ih  = (const float*)d_in[12];
  const float* b_hh  = (const float*)d_in[13];
  const float* W_dec = (const float*)d_in[14];
  const float* b_dec = (const float*)d_in[15];
  const float* h0    = (const float*)d_in[16];
  const float* c0    = (const float*)d_in[17];

  char* ws = (char*)d_ws;
  float*    bags = (float*)ws;                    // 16384*300*4  = 19,660,800
  _Float16* xh   = (_Float16*)(ws + 19660800);    // 16384*320*2  = 10,485,760
  _Float16* wih  = (_Float16*)(ws + 30146560);    // 256*320*2    = 163,840
  _Float16* whh  = (_Float16*)(ws + 30310400);    // 256*64*2     = 32,768
  _Float16* wdec = (_Float16*)(ws + 30343168);    // 16*64*2      = 2,048
  float*    gx   = (float*)(ws + 30345216);       // 16384*256*4  = 16,777,216
  float*    hsb  = (float*)(ws + 47122432);       // 16384*64*4   = 4,194,304

  float* dec = (float*)d_out;                     // [512,32,13]
  float* hT  = dec + SLEN * BSZ * NCLS;           // [32,64]
  float* cT  = hT + BSZ * HDIM;                   // [32,64]

  zero_f<<<4800, 256, 0, stream>>>(bags, ROWS * WDIM);
  scatter_bow<<<163840, 128, 0, stream>>>(wids, wseg, wtab, bags);
  pack_x<<<ROWS, 64, 0, stream>>>(spk, da, lvl, ulen, e_spk, e_da, e_lvl, bags, xh);
  conv_w<<<388, 256, 0, stream>>>(W_ih, W_hh, W_dec, wih, whh, wdec);
  gemm_gates<<<256, 256, 0, stream>>>(xh, wih, b_ih, b_hh, gx);
  lstm_seq<<<1, 256, 0, stream>>>(gx, whh, h0, c0, hsb, hT, cT);
  decode_k<<<128, 256, 0, stream>>>(hsb, wdec, b_dec, dec);
}
